// HapCL_44195213476240
// MI455X (gfx1250) — matrix operand order repaired and checked by reading, not compiled, not run on hardware
//
#include <hip/hip_runtime.h>
#include <cstdint>
#include <cstddef>

#define NBK   200000
#define NIT   50000
#define NTOT  250000
#define DD    64
#define HH    64
#define KQ    4
#define NNZE  2000000
#define BB    128
#define TLS   50
#define TLI   200

typedef __attribute__((ext_vector_type(16))) __bf16         v16bf;
typedef __attribute__((ext_vector_type(8)))  float          v8f;
typedef __attribute__((ext_vector_type(8)))  unsigned short v8u;
typedef __attribute__((ext_vector_type(16))) unsigned short v16u;

__device__ __forceinline__ unsigned short f2bf(float x) {
    unsigned int u = __float_as_uint(x);
    unsigned int r = (u + 0x7FFFu + ((u >> 16) & 1u)) >> 16;   // RNE
    return (unsigned short)r;
}
__device__ __forceinline__ float sigf(float x) { return 1.f / (1.f + __expf(-x)); }

__device__ __forceinline__ v16bf cat_bf(v8u lo, v8u hi) {
    v16u u;
#pragma unroll
    for (int i = 0; i < 8; i++) { u[i] = lo[i]; u[8 + i] = hi[i]; }
    return __builtin_bit_cast(v16bf, u);
}

// A-matrix fragment (16xK slice, 16-bit): lanes 0-15 -> M=lane, K = {0..7,16..23}+32*half
//                                         lanes 16-31 -> M=lane-16, K = {8..15,24..31}+32*half
__device__ __forceinline__ v16bf load_A(const unsigned short* base, int stride, int lane, int half) {
    int m  = lane & 15;
    int kb = ((lane & 16) ? 8 : 0) + half * 32;
    const unsigned short* p = base + (size_t)m * stride + kb;
    v8u lo = *(const v8u*)p;
    v8u hi = *(const v8u*)(p + 16);
    return cat_bf(lo, hi);
}
// B-matrix fragment (Kx16, 16-bit), stored as rows indexed by N (i.e. W[n][k] / items[n][k]):
// lanes 0-15 -> N=lane, K = {0..7,16..23}+32*half; lanes 16-31 -> N=lane-16, K = {8..15,24..31}+32*half
__device__ __forceinline__ v16bf load_B(const unsigned short* base, int stride, int lane, int half) {
    int n  = lane & 15;
    int kb = ((lane & 16) ? 8 : 0) + half * 32;
    const unsigned short* p = base + (size_t)n * stride + kb;
    v8u lo = *(const v8u*)p;
    v8u hi = *(const v8u*)(p + 16);
    return cat_bf(lo, hi);
}
__device__ __forceinline__ v8f wmma_bf(v16bf a, v16bf b, v8f c) {
    return __builtin_amdgcn_wmma_f32_16x16x32_bf16(false, a, false, b, (short)0, c, false, false);
}

// ---------------- Stage 1: LightGCN propagation ----------------
__global__ void __launch_bounds__(256) k_init(const float* __restrict__ eb, const float* __restrict__ ei,
                                              float* __restrict__ cur, float* __restrict__ acc) {
    int i = blockIdx.x * 256 + threadIdx.x;
    float v = (i < NBK * DD) ? eb[i] : ei[i - NBK * DD];
    cur[i] = v; acc[i] = v;
}

__global__ void __launch_bounds__(256) k_scatter(const float* __restrict__ cur, float* __restrict__ nxt,
                                                 const int* __restrict__ rows, const int* __restrict__ cols,
                                                 const float* __restrict__ vals) {
    int gid  = blockIdx.x * 256 + threadIdx.x;
    int e    = gid >> 5;
    int lane = gid & 31;
    if (e >= NNZE) return;
    int r = rows[e], c = cols[e];
    float v = vals[e];
    float2 d = ((const float2*)(cur + (size_t)c * DD))[lane];
    atomicAdd(&nxt[(size_t)r * DD + 2 * lane],     v * d.x);
    atomicAdd(&nxt[(size_t)r * DD + 2 * lane + 1], v * d.y);
}

__global__ void __launch_bounds__(256) k_accum(float* __restrict__ acc, const float* __restrict__ nxt) {
    int i = blockIdx.x * 256 + threadIdx.x;
    acc[i] += nxt[i];
}

__global__ void __launch_bounds__(256) k_itemsbf(const float* __restrict__ acc, unsigned short* __restrict__ itbf) {
    int i = blockIdx.x * 256 + threadIdx.x;           // NIT*DD
    itbf[i] = f2bf(0.25f * acc[(size_t)NBK * DD + i]);
}

// ---------------- Stage 2: gather + input transform (x = seq_emb @ W_int^T), bf16 out ----------------
__global__ void __launch_bounds__(256) k_xbf(const float* __restrict__ part, const int* __restrict__ seqs,
                                             const float* __restrict__ Wint, int T,
                                             unsigned short* __restrict__ Xout) {
    __shared__ float Wl[256 * DD];
    __shared__ float se[DD];
    int tid = threadIdx.x;
    for (int i = tid; i < 256 * DD; i += 256) Wl[i] = Wint[i];
    int b = blockIdx.x / T, t = blockIdx.x - b * T;
    if (tid < DD) {
        int idx = seqs[b * T + t];
        se[tid] = 0.25f * part[(size_t)idx * DD + tid];   // fold lightGCN 1/(L+1) scale here
    }
    __syncthreads();
    int d = tid >> 2, k = tid & 3;                        // W row = d*K + k == tid
    const float* wr = &Wl[tid * DD];
    float s = 0.f;
#pragma unroll
    for (int m = 0; m < DD; m++) s += wr[m] * se[m];
    Xout[(((size_t)(k * BB + b)) * T + t) * DD + d] = f2bf(s);
}

// ---------------- Stage 3: K*B parallel GRUs via bf16 WMMA ----------------
// 32 waves total (8 blocks x 4 waves); wave owns 16 sequences; 48 v_wmma per step.
__global__ void __launch_bounds__(128)
k_gru(const unsigned short* __restrict__ X, int T,
      const float* __restrict__ Wih, const float* __restrict__ Whh,
      const float* __restrict__ bih, const float* __restrict__ bhh,
      const int* __restrict__ slen, float* __restrict__ eout) {
    __shared__ unsigned short wih_s[192 * 72];   // padded rows: 144B -> conflict-free b128 reads
    __shared__ unsigned short whh_s[192 * 72];
    __shared__ float bihs[192], bhhs[192];
    __shared__ unsigned short h_s[4 * 16 * 72];  // per-wave h tile (bf16)

    const int tid = threadIdx.x;
    for (int i = tid; i < 192 * DD; i += 128) {
        int n = i >> 6, k = i & 63;
        wih_s[n * 72 + k] = f2bf(Wih[i]);
        whh_s[n * 72 + k] = f2bf(Whh[i]);
    }
    for (int i = tid; i < 192; i += 128) { bihs[i] = bih[i]; bhhs[i] = bhh[i]; }
    for (int i = tid; i < 4 * 16 * 72; i += 128) h_s[i] = 0;
    __syncthreads();

    const int lane  = tid & 31;
    const int wv    = tid >> 5;
    const int mbase = (blockIdx.x * 4 + wv) * 16;
    const int l15   = lane & 15;
    const int hi16  = (lane & 16) ? 8 : 0;
    unsigned short* hs = h_s + wv * (16 * 72);

    int lenr[8];
#pragma unroll
    for (int r = 0; r < 8; r++) lenr[r] = slen[(mbase + r + hi16) & (BB - 1)];

    v8f hacc[4];
#pragma unroll
    for (int j = 0; j < 4; j++)
#pragma unroll
        for (int i = 0; i < 8; i++) hacc[j][i] = 0.f;

    const unsigned short* xw = X + (size_t)mbase * T * DD;

    for (int t = 0; t < T; ++t) {
        v8f srz[8], gin[4], ghn[4];
#pragma unroll
        for (int nt = 0; nt < 8; nt++) {
            float bv = bihs[nt * 16 + l15] + bhhs[nt * 16 + l15];
#pragma unroll
            for (int i = 0; i < 8; i++) srz[nt][i] = bv;
        }
#pragma unroll
        for (int j = 0; j < 4; j++) {
            float bi_ = bihs[128 + j * 16 + l15];
            float bh_ = bhhs[128 + j * 16 + l15];
#pragma unroll
            for (int i = 0; i < 8; i++) { gin[j][i] = bi_; ghn[j][i] = bh_; }
        }
        const unsigned short* xb = xw + (size_t)t * DD;
#pragma unroll
        for (int half = 0; half < 2; ++half) {
            v16bf ax = load_A(xb, T * DD, lane, half);
            v16bf ah = load_A(hs, 72, lane, half);
#pragma unroll
            for (int nt = 0; nt < 8; nt++) {
                srz[nt] = wmma_bf(ax, load_B(wih_s + nt * 16 * 72, 72, lane, half), srz[nt]);
                srz[nt] = wmma_bf(ah, load_B(whh_s + nt * 16 * 72, 72, lane, half), srz[nt]);
            }
#pragma unroll
            for (int j = 0; j < 4; j++) {
                gin[j] = wmma_bf(ax, load_B(wih_s + (8 + j) * 16 * 72, 72, lane, half), gin[j]);
                ghn[j] = wmma_bf(ah, load_B(whh_s + (8 + j) * 16 * 72, 72, lane, half), ghn[j]);
            }
        }
        if (t + 1 < T) __builtin_prefetch(xb + DD + (size_t)l15 * T * DD, 0, 1);
#pragma unroll
        for (int j = 0; j < 4; j++) {
#pragma unroll
            for (int r = 0; r < 8; r++) {
                float rv = sigf(srz[j][r]);
                float zv = sigf(srz[4 + j][r]);
                float nv = tanhf(gin[j][r] + rv * ghn[j][r]);
                float hv = (1.f - zv) * nv + zv * hacc[j][r];
                hacc[j][r] = hv;
                hs[(r + hi16) * 72 + j * 16 + l15] = f2bf(hv);
                if (t == lenr[r] - 1) {
                    int s = mbase + r + hi16;
                    int b = s & (BB - 1), kq = s >> 7;
                    eout[((size_t)(b * KQ + kq)) * HH + j * 16 + l15] = hv;  // e[b,k,h]
                }
            }
        }
    }
}

// ---------------- Stage 4: attention collapse -> per-batch query vector ----------------
__global__ void k_merge(const float* __restrict__ e, const float* __restrict__ wm, float* __restrict__ q) {
    int b = threadIdx.x;
    if (b >= BB) return;
    const float* eb = e + (size_t)b * KQ * HH;
    float dot[KQ][KQ];
    for (int k = 0; k < KQ; k++)
        for (int j = 0; j < KQ; j++) {
            float s = 0.f;
            for (int h = 0; h < HH; h++) s += eb[k * HH + h] * eb[j * HH + h];
            dot[k][j] = s;
        }
    float mcoef[KQ];
    for (int k = 0; k < KQ; k++) {
        float mx = dot[k][0];
        for (int j = 1; j < KQ; j++) mx = fmaxf(mx, dot[k][j]);
        float p[KQ], den = 0.f;
        for (int j = 0; j < KQ; j++) { p[j] = __expf(dot[k][j] - mx); den += p[j]; }
        float m = 0.f;
        for (int j = 0; j < KQ; j++) m += (p[j] / den) * wm[j];
        mcoef[k] = m;
    }
    for (int h = 0; h < HH; h++) {
        float s = 0.f;
        for (int k = 0; k < KQ; k++) s += mcoef[k] * eb[k * HH + h];
        q[b * HH + h] += s;   // branches accumulate (sequential launches)
    }
}

__global__ void __launch_bounds__(256) k_qbf(const float* __restrict__ q, unsigned short* __restrict__ qbf) {
    int i = blockIdx.x * 256 + threadIdx.x;   // BB*HH
    qbf[i] = f2bf(q[i]);
}

// ---------------- Stage 5: out = sigmoid(Q @ items^T), bf16 WMMA ----------------
__global__ void __launch_bounds__(256)
k_logits(const unsigned short* __restrict__ qbf, const unsigned short* __restrict__ itbf,
         float* __restrict__ out) {
    int lane = threadIdx.x & 31, w = threadIdx.x >> 5;   // 8 waves = 8 M-tiles (B=128)
    int nt = blockIdx.x;                                  // N-tile over items
    v8f c;
#pragma unroll
    for (int i = 0; i < 8; i++) c[i] = 0.f;
    const unsigned short* abase = qbf  + (size_t)w  * 16 * DD;
    const unsigned short* bbase = itbf + (size_t)nt * 16 * DD;
#pragma unroll
    for (int half = 0; half < 2; half++) {
        v16bf a = load_A(abase, DD, lane, half);
        v16bf b = load_B(bbase, DD, lane, half);
        c = wmma_bf(a, b, c);
    }
    int i    = nt * 16 + (lane & 15);
    int hi16 = (lane & 16) ? 8 : 0;
#pragma unroll
    for (int r = 0; r < 8; r++) {
        int b = w * 16 + r + hi16;
        out[(size_t)b * NIT + i] = sigf(c[r]);
    }
}

extern "C" void kernel_launch(void* const* d_in, const int* in_sizes, int n_in,
                              void* d_out, int out_size, void* d_ws, size_t ws_size,
                              hipStream_t stream) {
    (void)in_sizes; (void)n_in; (void)out_size; (void)ws_size;
    const float* emb_basket = (const float*)d_in[0];
    const float* emb_item   = (const float*)d_in[1];
    const float* adj_vals   = (const float*)d_in[2];
    const float* W_bseq     = (const float*)d_in[3];
    const float* W_iseq     = (const float*)d_in[4];
    const float* gWih = (const float*)d_in[5];
    const float* gWhh = (const float*)d_in[6];
    const float* gbih = (const float*)d_in[7];
    const float* gbhh = (const float*)d_in[8];
    const float* iWih = (const float*)d_in[9];
    const float* iWhh = (const float*)d_in[10];
    const float* ibih = (const float*)d_in[11];
    const float* ibhh = (const float*)d_in[12];
    const float* Wmb  = (const float*)d_in[13];
    const float* Wmi  = (const float*)d_in[14];
    const int* adj_rows = (const int*)d_in[15];
    const int* adj_cols = (const int*)d_in[16];
    const int* bseq     = (const int*)d_in[17];
    const int* bseq_len = (const int*)d_in[18];
    const int* iseq     = (const int*)d_in[19];
    const int* iseq_len = (const int*)d_in[20];
    float* out = (float*)d_out;

    char* w = (char*)d_ws;
    float* bufA = (float*)w;                       // 64 MB (N*64 f32)
    float* bufB = (float*)(w + 64000000);          // 64 MB
    float* acc  = (float*)(w + 128000000);         // 64 MB (stays live)
    // region of bufA/bufB is reused after the graph stage:
    unsigned short* Xi   = (unsigned short*)w;                 // 13.1 MB
    unsigned short* Xb   = (unsigned short*)(w + 13107200);    // 3.3 MB
    unsigned short* itbf = (unsigned short*)(w + 16384000);    // 6.4 MB
    float* e_b = (float*)(w + 22784000);
    float* e_i = (float*)(w + 22915072);
    float* q   = (float*)(w + 23046144);
    unsigned short* qbf = (unsigned short*)(w + 23078912);

    // Stage 1: 3-layer propagation, acc = sum of layers (x0..x3)
    k_init<<<62500, 256, 0, stream>>>(emb_basket, emb_item, bufA, acc);
    float* cur = bufA; float* nxt = bufB;
    for (int l = 0; l < 3; l++) {
        hipMemsetAsync(nxt, 0, (size_t)NTOT * DD * 4, stream);
        k_scatter<<<NNZE / 8, 256, 0, stream>>>(cur, nxt, adj_rows, adj_cols, adj_vals);
        k_accum<<<62500, 256, 0, stream>>>(acc, nxt);
        float* tmp = cur; cur = nxt; nxt = tmp;
    }

    // Stage 2: items -> bf16 (scaled by 0.25); gather + input transforms
    k_itemsbf<<<12500, 256, 0, stream>>>(acc, itbf);
    k_xbf<<<BB * TLS, 256, 0, stream>>>(acc,                    bseq, W_bseq, TLS, Xb);
    k_xbf<<<BB * TLI, 256, 0, stream>>>(acc + (size_t)NBK * DD, iseq, W_iseq, TLI, Xi);

    // Stage 3: GRUs (WMMA)
    k_gru<<<8, 128, 0, stream>>>(Xb, TLS, gWih, gWhh, gbih, gbhh, bseq_len, e_b);
    k_gru<<<8, 128, 0, stream>>>(Xi, TLI, iWih, iWhh, ibih, ibhh, iseq_len, e_i);

    // Stage 4: attention collapse, both branches accumulate into q
    hipMemsetAsync(q, 0, BB * HH * 4, stream);
    k_merge<<<1, 128, 0, stream>>>(e_b, Wmb, q);
    k_merge<<<1, 128, 0, stream>>>(e_i, Wmi, q);
    k_qbf<<<32, 256, 0, stream>>>(q, qbf);

    // Stage 5: final GEMM + sigmoid (WMMA)
    k_logits<<<NIT / 16, 256, 0, stream>>>(qbf, itbf, out);
}
